// model_49752901157476
// MI455X (gfx1250) — compile-verified, matrix operand order checked by reading, not run on hardware
//
#include <hip/hip_runtime.h>

// ---------------------------------------------------------------------------
// CDNA5 (gfx1250, wave32) implementation of the seq2seq LSTM+attention model.
// All GEMMs use v_wmma_f32_16x16x32_bf16 (bf16 inputs, fp32 accumulate).
// Logits kernel stages gen_W slabs into LDS via the Tensor Data Mover when
// the toolchain exposes it (guarded), falling back to vector copies.
// ---------------------------------------------------------------------------

typedef __attribute__((ext_vector_type(16))) __bf16 v16bf;
typedef __attribute__((ext_vector_type(8)))  float  v8f;
typedef unsigned int u32x4 __attribute__((ext_vector_type(4)));
typedef int          i32x4 __attribute__((ext_vector_type(4)));
typedef int          i32x8 __attribute__((ext_vector_type(8)));

#if defined(__has_builtin)
#if __has_builtin(__builtin_amdgcn_tensor_load_to_lds) && \
    __has_builtin(__builtin_amdgcn_s_wait_tensorcnt)
#define USE_TDM 1
#endif
#endif

#define DEVINL __device__ __forceinline__

// round-to-nearest-even f32 -> bf16 (bit pattern in ushort)
DEVINL unsigned short f2bf(float x) {
  unsigned int u = __float_as_uint(x);
  unsigned int r = (u + 0x7FFFu + ((u >> 16) & 1u)) >> 16;
  return (unsigned short)r;
}

// Load one WMMA 16x32 bf16 fragment (A layout; B columns use the same pattern
// with the tile's row base = W + n0*ldw).  Per ISA 7.12.2: lane&15 selects the
// matrix row (A) / column (B), lane>>4 selects the K half, each VGPR holds a
// packed pair of consecutive-K bf16 values -> two b128 loads per lane.
DEVINL v16bf load_frag(const unsigned short* base, int ld, int k0) {
  int lane = (int)(threadIdx.x & 31u);
  int r    = lane & 15;
  int half = lane >> 4;
  union { v16bf v; unsigned int u[8]; } f;
  const unsigned short* row = base + r * ld + k0 + 8 * half;
#pragma unroll
  for (int j = 0; j < 8; ++j) {
    int kk = (j < 4) ? (2 * j) : (16 + 2 * (j - 4));
    f.u[j] = *(const unsigned int*)(row + kk);
  }
  return f.v;
}

DEVINL v8f wmma_bf16(const v16bf& a, const v16bf& b, const v8f& c) {
  return __builtin_amdgcn_wmma_f32_16x16x32_bf16(false, a, false, b,
                                                 (short)0, c, false, false);
}

// acc += A[16xK] * W[n-tile rows, K]^T   (A, W bf16; K % 32 == 0)
DEVINL void wmma_loop(v8f& acc, const unsigned short* A, int lda,
                      const unsigned short* Wt, int ldw, int K) {
  for (int k0 = 0; k0 < K; k0 += 32) {
    v16bf a = load_frag(A, lda, k0);
    v16bf b = load_frag(Wt, ldw, k0);
    acc = wmma_bf16(a, b, acc);
  }
}

DEVINL float sigm(float x) { return 1.0f / (1.0f + __expf(-x)); }

// ---------------------------------------------------------------------------
// small utility kernels
// ---------------------------------------------------------------------------
__global__ void cvt_k(const float* __restrict__ src,
                      unsigned short* __restrict__ dst, long n) {
  long i = (long)blockIdx.x * blockDim.x + threadIdx.x;
  if (i < n) dst[i] = f2bf(src[i]);
}

// xemb[(s*16+b)*512+h] = bf16(encemb[inp[b,s], h])
__global__ void gather_enc_k(const int* __restrict__ inp,
                             const float* __restrict__ emb,
                             unsigned short* __restrict__ dst) {
  long i = (long)blockIdx.x * blockDim.x + threadIdx.x;
  if (i >= (long)128 * 16 * 512) return;
  int h = (int)(i & 511);
  long sb = i >> 9;
  int b = (int)(sb & 15);
  int s = (int)(sb >> 4);
  int tok = inp[b * 128 + s];
  dst[sb * 512 + h] = f2bf(emb[(long)tok * 512 + h]);
}

// demb[(t*16+b)*512+h] = bf16(decemb[prev, h]), prev = t==0 ? SOS : out[b,t-1]
__global__ void gather_dec_k(const int* __restrict__ outtok,
                             const float* __restrict__ emb,
                             unsigned short* __restrict__ dst) {
  long i = (long)blockIdx.x * blockDim.x + threadIdx.x;
  if (i >= (long)128 * 16 * 512) return;
  int h = (int)(i & 511);
  long tb = i >> 9;
  int b = (int)(tb & 15);
  int t = (int)(tb >> 4);
  int tok = (t == 0) ? 3 : outtok[b * 128 + (t - 1)];
  dst[tb * 512 + h] = f2bf(emb[(long)tok * 512 + h]);
}

// catenc[(b*64+k)*512+h] = catemb[catvec[b,k], h]   (fp32)
__global__ void gather_cat_k(const int* __restrict__ catvec,
                             const float* __restrict__ emb,
                             float* __restrict__ dst) {
  long i = (long)blockIdx.x * blockDim.x + threadIdx.x;
  if (i >= (long)16 * 64 * 512) return;
  int h = (int)(i & 511);
  long bk = i >> 9;
  int idx = catvec[bk];
  dst[bk * 512 + h] = emb[(long)idx * 512 + h];
}

// ---------------------------------------------------------------------------
// generic bf16 WMMA GEMM:  C[M,N](f32) = A[M,K](bf16) @ W[N,K](bf16)^T (+bias)
// one wave per 16x64 output slab (A fragment reused across 4 WMMA tiles)
// ---------------------------------------------------------------------------
__global__ void gemm_bf16_k(const unsigned short* __restrict__ A, int lda,
                            const unsigned short* __restrict__ W, int ldw,
                            float* __restrict__ C, int ldc,
                            const float* __restrict__ bias,
                            int Mtiles, int Ngrp, int K) {
  long wid = ((long)blockIdx.x * blockDim.x + threadIdx.x) >> 5;
  if (wid >= (long)Mtiles * Ngrp) return;
  int tm = (int)(wid % Mtiles), tg = (int)(wid / Mtiles);
  int m0 = tm * 16, n0 = tg * 64;
  const unsigned short* Arow = A + (long)m0 * lda;
  v8f acc[4] = {};
  for (int k0 = 0; k0 < K; k0 += 32) {
    v16bf a = load_frag(Arow, lda, k0);
#pragma unroll
    for (int u = 0; u < 4; ++u) {
      v16bf b = load_frag(W + (long)(n0 + u * 16) * ldw, ldw, k0);
      acc[u] = wmma_bf16(a, b, acc[u]);
    }
  }
  int lane = (int)(threadIdx.x & 31u);
  int n = lane & 15, half = lane >> 4;
#pragma unroll
  for (int u = 0; u < 4; ++u) {
    int nn = n0 + u * 16 + n;
    float badd = bias ? bias[nn] : 0.0f;
#pragma unroll
    for (int r = 0; r < 8; ++r) {
      int m = m0 + r + 8 * half;
      C[(long)m * ldc + nn] = acc[u][r] + badd;
    }
  }
}

// ---------------------------------------------------------------------------
// vocab logits: out[b,t,v] = ops[t,b,:] . gen_W[v,:] + gen_b[v]
// Block = one 16-vocab n-tile x 64 t-steps. gen_W slab (16x512 bf16 = 16KB)
// staged to LDS (TDM when available), then 8 waves x 8 t-tiles each reuse the
// B fragments from LDS across 8 WMMAs.
// ---------------------------------------------------------------------------
__global__ __launch_bounds__(256) void logits_k(
    const unsigned short* __restrict__ opsb,
    const unsigned short* __restrict__ genb,
    const float* __restrict__ gbias,
    float* __restrict__ outp) {
  __shared__ unsigned short Wt[16 * 512];
  int tid = (int)threadIdx.x;
  int blk = (int)blockIdx.x;          // 2000 n-tiles * 2 t-groups
  int ngrp = blk >> 1;
  int tgrp = blk & 1;
  int n0 = ngrp * 16;
  const unsigned short* Wsrc = genb + (long)n0 * 512;

#ifdef USE_TDM
  if ((tid >> 5) == 0) {
    unsigned long long ga = (unsigned long long)Wsrc;
    unsigned int la = (unsigned int)(unsigned long long)(&Wt[0]);  // LDS byte addr
    u32x4 g0;
    g0[0] = 1u;                                            // count=1 (valid user D#)
    g0[1] = la;                                            // lds_addr
    g0[2] = (unsigned int)(ga & 0xFFFFFFFFu);              // global_addr[31:0]
    g0[3] = (unsigned int)((ga >> 32) & 0x01FFFFFFu) | (2u << 30);  // addr hi | type=2
    i32x8 g1;
    g1[0] = (1 << 16);              // workgroup_mask=0, data_size=1 (2 bytes)
    g1[1] = (int)(512u << 16);      // tensor_dim0 = 512 (low 16 bits at [63:48])
    g1[2] = (int)(16u << 16);       // tensor_dim0 hi=0 | tensor_dim1 = 16
    g1[3] = (int)(512u << 16);      // tensor_dim1 hi=0 | tile_dim0 = 512
    g1[4] = 16;                     // tile_dim1 = 16, tile_dim2 = 0
    g1[5] = 512;                    // tensor_dim0_stride (elements)
    g1[6] = 0;
    g1[7] = 0;
    i32x4 gz = {0, 0, 0, 0};
#if __clang_major__ >= 23
    i32x8 gz8 = {0, 0, 0, 0, 0, 0, 0, 0};
    __builtin_amdgcn_tensor_load_to_lds(g0, g1, gz, gz, gz8, 0);
#else
    __builtin_amdgcn_tensor_load_to_lds(g0, g1, gz, gz, 0);
#endif
    __builtin_amdgcn_s_wait_tensorcnt(0);
  }
  __syncthreads();
#else
  {
    const uint4* s4 = (const uint4*)Wsrc;
    uint4* d4 = (uint4*)Wt;
    for (int i = tid; i < 1024; i += 256) d4[i] = s4[i];
    __syncthreads();
  }
#endif

  int wave = tid >> 5;
  int tbase = tgrp * 64 + wave * 8;
  v8f acc[8] = {};
  for (int k0 = 0; k0 < 512; k0 += 32) {
    v16bf b = load_frag(Wt, 512, k0);   // from LDS
#pragma unroll
    for (int u = 0; u < 8; ++u) {
      v16bf a = load_frag(opsb + (long)(tbase + u) * 16 * 512, 512, k0);
      acc[u] = wmma_bf16(a, b, acc[u]);
    }
  }
  int lane = (int)(threadIdx.x & 31u);
  int n = lane & 15, half = lane >> 4;
  float badd = gbias[n0 + n];
#pragma unroll
  for (int u = 0; u < 8; ++u) {
    int t = tbase + u;
#pragma unroll
    for (int r = 0; r < 8; ++r) {
      int b = r + 8 * half;  // batch row
      outp[((long)b * 128 + t) * 32000 + n0 + n] = acc[u][r] + badd;
    }
  }
}

// ---------------------------------------------------------------------------
// encoder LSTM scan, one direction per block (grid = 2), 512 threads/block.
// Gpre = x@Wih^T precomputed for all steps; per step: g = Gpre[t] + h@Whh^T + b
// h: [16,256] bf16 in LDS, c: [16,256] f32 in LDS, gates g: [16,1024] f32 LDS.
// ---------------------------------------------------------------------------
__global__ __launch_bounds__(512) void enc_lstm_k(
    const float* __restrict__ Gpre0, const float* __restrict__ Gpre1,
    const unsigned short* __restrict__ WhhB,  // + dir*1024*256
    const float* __restrict__ bias2,          // + dir*1024
    unsigned short* __restrict__ ybf,         // layer0: x1 bf16 [2048,512] (or null)
    float* __restrict__ yf32,                 // layer1: enc f32 [16,128,512] (or null)
    float* __restrict__ hfin, float* __restrict__ cfin)  // [16*512] this layer
{
  extern __shared__ char smem[];
  float* g = (float*)smem;                                        // 16*1024 f32
  float* c = (float*)(smem + 16 * 1024 * 4);                      // 16*256 f32
  unsigned short* hb = (unsigned short*)(smem + 16 * 1024 * 4 + 16 * 256 * 4);

  int dir = blockIdx.x;
  const float* Gpre = dir ? Gpre1 : Gpre0;
  const unsigned short* Whh = WhhB + (long)dir * 1024 * 256;
  const float* bias = bias2 + dir * 1024;
  int tid = (int)threadIdx.x, wave = tid >> 5, lane = tid & 31;
  int n_ = lane & 15, half = lane >> 4;

  for (int i = tid; i < 16 * 256; i += 512) { c[i] = 0.0f; hb[i] = 0; }
  __syncthreads();

  for (int t = 0; t < 128; ++t) {
    int s = dir ? (127 - t) : t;
    // ---- gates: 64 n-tiles over 16 waves ----
    for (int tt = 0; tt < 4; ++tt) {
      int n0 = (wave * 4 + tt) * 16;
      v8f acc;
#pragma unroll
      for (int r = 0; r < 8; ++r) {
        int m = r + 8 * half;
        acc[r] = Gpre[((long)(s * 16 + m)) * 1024 + n0 + n_] + bias[n0 + n_];
      }
      wmma_loop(acc, hb, 256, Whh + (long)n0 * 256, 256, 256);
#pragma unroll
      for (int r = 0; r < 8; ++r) {
        int m = r + 8 * half;
        g[m * 1024 + n0 + n_] = acc[r];
      }
    }
    __syncthreads();
    // ---- cell update (torch gate order i,f,g,o) ----
    for (int i = tid; i < 16 * 256; i += 512) {
      int b = i >> 8, j = i & 255;
      float gi = g[b * 1024 + j];
      float gf = g[b * 1024 + 256 + j];
      float gg = g[b * 1024 + 512 + j];
      float go = g[b * 1024 + 768 + j];
      float cn = sigm(gf) * c[i] + sigm(gi) * tanhf(gg);
      float hn = sigm(go) * tanhf(cn);
      c[i] = cn;
      hb[i] = f2bf(hn);
      if (ybf)  ybf[((long)(s * 16 + b)) * 512 + dir * 256 + j] = f2bf(hn);
      if (yf32) yf32[((long)b * 128 + s) * 512 + dir * 256 + j] = hn;
      if (t == 127) {
        hfin[b * 512 + dir * 256 + j] = hn;
        cfin[b * 512 + dir * 256 + j] = cn;
      }
    }
    __syncthreads();
  }
}

// ---------------------------------------------------------------------------
// decoder: single persistent block (512 threads, 16 waves), T=128 steps.
// ---------------------------------------------------------------------------
__global__ __launch_bounds__(512) void dec_k(
    const float* __restrict__ preDt,            // [2048,2048] demb@Wih0[:, :512]^T
    const unsigned short* __restrict__ Wih0b,   // [2048,1024] bf16
    const unsigned short* __restrict__ Wih1b,   // [2048,512]
    const unsigned short* __restrict__ Whh0b,   // [2048,512]
    const unsigned short* __restrict__ Whh1b,   // [2048,512]
    const float* __restrict__ dec_b,            // [2,2048]
    const unsigned short* __restrict__ lininb,  // [512,512]
    const unsigned short* __restrict__ mixb,    // [512,1024] (cols 512.. = decout part)
    const float* __restrict__ catq_base,        // [16,512]
    const unsigned short* __restrict__ linoutb, // [512,1536]
    const float* __restrict__ enc,              // [16,128,512]
    const float* __restrict__ catenc,           // [16,64,512]
    const float* __restrict__ hinit,            // [2*16*512]
    const float* __restrict__ cinit,            // [2*16*512]
    unsigned short* __restrict__ opsb,          // [2048,512] bf16 out
    float* __restrict__ attn_out)               // [16,128,64]
{
  extern __shared__ char smem[];
  float* g            = (float*)smem;                       // 16*2048 f32 (gates)
  float* c0           = (float*)(smem + 131072);            // 16*512
  float* c1           = (float*)(smem + 163840);            // 16*512
  unsigned short* h0b = (unsigned short*)(smem + 196608);   // 16*512 bf16
  unsigned short* h1b = (unsigned short*)(smem + 212992);   // 16*512 bf16
  unsigned short* opb = (unsigned short*)(smem + 229376);   // 16*512 bf16
  unsigned short* catb= (unsigned short*)(smem + 245760);   // 16*1536 bf16 concat
  // aliases inside the (free-between-phases) gate region:
  float* qf  = (float*)smem;                  // 16*512
  float* cqf = (float*)(smem + 32768);        // 16*512
  float* sc  = (float*)(smem + 65536);        // 16*128
  float* cw  = (float*)(smem + 73728);        // 16*64

  int tid = (int)threadIdx.x, wave = tid >> 5, lane = tid & 31;
  int n_ = lane & 15, half = lane >> 4;

  for (int i = tid; i < 16 * 512; i += 512) {
    c0[i] = cinit[i];
    c1[i] = cinit[8192 + i];
    h0b[i] = f2bf(hinit[i]);
    h1b[i] = f2bf(hinit[8192 + i]);
    opb[i] = 0;
  }
  __syncthreads();

  for (int t = 0; t < 128; ++t) {
    // ---- layer0 gates: 128 tiles / 16 waves ----
    for (int tt = 0; tt < 8; ++tt) {
      int n0 = (wave * 8 + tt) * 16;
      v8f acc;
#pragma unroll
      for (int r = 0; r < 8; ++r) {
        int m = r + 8 * half;
        acc[r] = preDt[((long)(t * 16 + m)) * 2048 + n0 + n_] + dec_b[n0 + n_];
      }
      wmma_loop(acc, opb, 512, Wih0b + (long)n0 * 1024 + 512, 1024, 512);
      wmma_loop(acc, h0b, 512, Whh0b + (long)n0 * 512, 512, 512);
#pragma unroll
      for (int r = 0; r < 8; ++r) {
        int m = r + 8 * half;
        g[m * 2048 + n0 + n_] = acc[r];
      }
    }
    __syncthreads();
    for (int i = tid; i < 16 * 512; i += 512) {   // layer0 cell
      int b = i >> 9, j = i & 511;
      float gi = g[b * 2048 + j], gf = g[b * 2048 + 512 + j];
      float gg = g[b * 2048 + 1024 + j], go = g[b * 2048 + 1536 + j];
      float cn = sigm(gf) * c0[i] + sigm(gi) * tanhf(gg);
      float hn = sigm(go) * tanhf(cn);
      c0[i] = cn;
      h0b[i] = f2bf(hn);
    }
    __syncthreads();
    // ---- layer1 gates ----
    for (int tt = 0; tt < 8; ++tt) {
      int n0 = (wave * 8 + tt) * 16;
      v8f acc;
#pragma unroll
      for (int r = 0; r < 8; ++r) acc[r] = dec_b[2048 + n0 + n_];
      wmma_loop(acc, h0b, 512, Wih1b + (long)n0 * 512, 512, 512);
      wmma_loop(acc, h1b, 512, Whh1b + (long)n0 * 512, 512, 512);
#pragma unroll
      for (int r = 0; r < 8; ++r) {
        int m = r + 8 * half;
        g[m * 2048 + n0 + n_] = acc[r];
      }
    }
    __syncthreads();
    for (int i = tid; i < 16 * 512; i += 512) {   // layer1 cell, decout -> concat mid
      int b = i >> 9, j = i & 511;
      float gi = g[b * 2048 + j], gf = g[b * 2048 + 512 + j];
      float gg = g[b * 2048 + 1024 + j], go = g[b * 2048 + 1536 + j];
      float cn = sigm(gf) * c1[i] + sigm(gi) * tanhf(gg);
      float hn = sigm(go) * tanhf(cn);
      c1[i] = cn;
      unsigned short hv = f2bf(hn);
      h1b[i] = hv;
      catb[b * 1536 + 512 + j] = hv;
    }
    __syncthreads();
    // ---- q = decout@linin^T ; catq = catq_base + decout@mix[:,512:]^T ----
    for (int tt = 0; tt < 4; ++tt) {
      int job = wave * 4 + tt;  // 0..63
      if (job < 32) {
        int n0 = job * 16;
        v8f acc = {};
        wmma_loop(acc, h1b, 512, lininb + (long)n0 * 512, 512, 512);
#pragma unroll
        for (int r = 0; r < 8; ++r) {
          int m = r + 8 * half;
          qf[m * 512 + n0 + n_] = acc[r];
        }
      } else {
        int n0 = (job - 32) * 16;
        v8f acc;
#pragma unroll
        for (int r = 0; r < 8; ++r) {
          int m = r + 8 * half;
          acc[r] = catq_base[m * 512 + n0 + n_];
        }
        wmma_loop(acc, h1b, 512, mixb + (long)n0 * 1024 + 512, 1024, 512);
#pragma unroll
        for (int r = 0; r < 8; ++r) {
          int m = r + 8 * half;
          cqf[m * 512 + n0 + n_] = acc[r];
        }
      }
    }
    __syncthreads();
    // ---- attention scores ----
    for (int i = tid; i < 2048; i += 512) {       // enc scores [16,128]
      int b = i >> 7, s = i & 127;
      const float* er = enc + ((long)b * 128 + s) * 512;
      const float* qr = qf + b * 512;
      float acc = 0.0f;
      for (int h = 0; h < 512; ++h) acc += er[h] * qr[h];
      sc[i] = acc;
    }
    for (int i = tid; i < 1024; i += 512) {       // cat scores [16,64]
      int b = i >> 6, k = i & 63;
      const float* er = catenc + ((long)b * 64 + k) * 512;
      const float* qr = cqf + b * 512;
      float acc = 0.0f;
      for (int h = 0; h < 512; ++h) acc += er[h] * qr[h];
      cw[i] = acc;
    }
    __syncthreads();
    // ---- softmax (wave w handles batch b=w) ----
    {
      int b = wave;
      float v0 = sc[b * 128 + lane], v1 = sc[b * 128 + 32 + lane];
      float v2 = sc[b * 128 + 64 + lane], v3 = sc[b * 128 + 96 + lane];
      float mx = fmaxf(fmaxf(v0, v1), fmaxf(v2, v3));
      for (int o = 16; o > 0; o >>= 1) mx = fmaxf(mx, __shfl_xor(mx, o, 32));
      float e0 = __expf(v0 - mx), e1 = __expf(v1 - mx);
      float e2 = __expf(v2 - mx), e3 = __expf(v3 - mx);
      float ssum = e0 + e1 + e2 + e3;
      for (int o = 16; o > 0; o >>= 1) ssum += __shfl_xor(ssum, o, 32);
      float inv = 1.0f / ssum;
      sc[b * 128 + lane] = e0 * inv;
      sc[b * 128 + 32 + lane] = e1 * inv;
      sc[b * 128 + 64 + lane] = e2 * inv;
      sc[b * 128 + 96 + lane] = e3 * inv;

      float u0 = cw[b * 64 + lane], u1 = cw[b * 64 + 32 + lane];
      float mx2 = fmaxf(u0, u1);
      for (int o = 16; o > 0; o >>= 1) mx2 = fmaxf(mx2, __shfl_xor(mx2, o, 32));
      float f0 = __expf(u0 - mx2), f1 = __expf(u1 - mx2);
      float s2 = f0 + f1;
      for (int o = 16; o > 0; o >>= 1) s2 += __shfl_xor(s2, o, 32);
      float inv2 = 1.0f / s2;
      f0 *= inv2; f1 *= inv2;
      cw[b * 64 + lane] = f0;
      cw[b * 64 + 32 + lane] = f1;
      attn_out[((long)b * 128 + t) * 64 + lane] = f0;
      attn_out[((long)b * 128 + t) * 64 + 32 + lane] = f1;
    }
    __syncthreads();
    // ---- context vectors -> concat buffer (bf16) ----
    for (int i = tid; i < 16 * 512; i += 512) {
      int b = i >> 9, h = i & 511;
      const float* wr = sc + b * 128;
      const float* er = enc + ((long)b * 128) * 512 + h;
      float acc = 0.0f;
      for (int s = 0; s < 128; ++s) acc += wr[s] * er[s * 512];
      catb[b * 1536 + h] = f2bf(acc);
      const float* wr2 = cw + b * 64;
      const float* er2 = catenc + ((long)b * 64) * 512 + h;
      float acc2 = 0.0f;
      for (int k = 0; k < 64; ++k) acc2 += wr2[k] * er2[k * 512];
      catb[b * 1536 + 1024 + h] = f2bf(acc2);
    }
    __syncthreads();
    // ---- opn = tanh(concat @ linout^T) ----
    for (int tt = 0; tt < 2; ++tt) {
      int n0 = (wave * 2 + tt) * 16;
      v8f acc = {};
      wmma_loop(acc, catb, 1536, linoutb + (long)n0 * 1536, 1536, 1536);
#pragma unroll
      for (int r = 0; r < 8; ++r) {
        int m = r + 8 * half;
        float o = tanhf(acc[r]);
        unsigned short ov = f2bf(o);
        opb[m * 512 + n0 + n_] = ov;
        opsb[((long)(t * 16 + m)) * 512 + n0 + n_] = ov;
      }
    }
    __syncthreads();
  }
}

// ---------------------------------------------------------------------------
// host orchestration
// ---------------------------------------------------------------------------
extern "C" void kernel_launch(void* const* d_in, const int* in_sizes, int n_in,
                              void* d_out, int out_size, void* d_ws, size_t ws_size,
                              hipStream_t stream) {
  (void)in_sizes; (void)out_size; (void)ws_size;
  if (n_in < 19) return;
  const int*   inp      = (const int*)d_in[0];
  const int*   catvec   = (const int*)d_in[1];
  const int*   outtok   = (const int*)d_in[2];
  const float* encemb   = (const float*)d_in[3];
  const float* decemb   = (const float*)d_in[4];
  const float* catemb   = (const float*)d_in[5];
  const float* enc_Wih  = (const float*)d_in[6];
  const float* enc_Whh  = (const float*)d_in[7];
  const float* enc_b    = (const float*)d_in[8];
  const float* dec_Wih0 = (const float*)d_in[9];
  const float* dec_Wih1 = (const float*)d_in[10];
  const float* dec_Whh  = (const float*)d_in[11];
  const float* dec_bi   = (const float*)d_in[12];
  const float* lininW   = (const float*)d_in[13];
  const float* linoutW  = (const float*)d_in[14];
  const float* stepdW   = (const float*)d_in[15];
  const float* mixW     = (const float*)d_in[16];
  const float* genW     = (const float*)d_in[17];
  const float* genb     = (const float*)d_in[18];

  char* ws = (char*)d_ws;
  size_t off = 0;
  auto alloc = [&](size_t bytes) -> char* {
    char* p = ws + off;
    off = (off + bytes + 255) & ~(size_t)255;
    return p;
  };
  unsigned short* xemb    = (unsigned short*)alloc((size_t)2048 * 512 * 2);
  unsigned short* x1      = (unsigned short*)alloc((size_t)2048 * 512 * 2);
  float*          encf    = (float*)alloc((size_t)16 * 128 * 512 * 4);
  float*          Gp0     = (float*)alloc((size_t)2048 * 1024 * 4);
  float*          Gp1     = (float*)alloc((size_t)2048 * 1024 * 4);
  unsigned short* WihB    = (unsigned short*)alloc((size_t)4 * 1024 * 512 * 2);
  unsigned short* WhhB    = (unsigned short*)alloc((size_t)4 * 1024 * 256 * 2);
  float*          hfin    = (float*)alloc((size_t)2 * 16 * 512 * 4);
  float*          cfin    = (float*)alloc((size_t)2 * 16 * 512 * 4);
  unsigned short* hflatb  = (unsigned short*)alloc((size_t)16384 * 2);
  unsigned short* stepdB  = (unsigned short*)alloc((size_t)512 * 1024 * 2);
  float*          e2c     = (float*)alloc((size_t)16 * 512 * 4);
  unsigned short* e2cb    = (unsigned short*)alloc((size_t)16 * 512 * 2);
  unsigned short* mixB    = (unsigned short*)alloc((size_t)512 * 1024 * 2);
  float*          catqb   = (float*)alloc((size_t)16 * 512 * 4);
  float*          catencf = (float*)alloc((size_t)16 * 64 * 512 * 4);
  unsigned short* demb    = (unsigned short*)alloc((size_t)2048 * 512 * 2);
  unsigned short* Wih0B   = (unsigned short*)alloc((size_t)2048 * 1024 * 2);
  unsigned short* Wih1B   = (unsigned short*)alloc((size_t)2048 * 512 * 2);
  unsigned short* WhhdB   = (unsigned short*)alloc((size_t)2 * 2048 * 512 * 2);
  float*          preDt   = (float*)alloc((size_t)2048 * 2048 * 4);
  unsigned short* lininB  = (unsigned short*)alloc((size_t)512 * 512 * 2);
  unsigned short* linoutB = (unsigned short*)alloc((size_t)512 * 1536 * 2);
  unsigned short* opsB    = (unsigned short*)alloc((size_t)2048 * 512 * 2);
  unsigned short* genB    = (unsigned short*)alloc((size_t)32000 * 512 * 2);

  auto cvt = [&](const float* s, unsigned short* d, long n) {
    int blocks = (int)((n + 255) / 256);
    cvt_k<<<blocks, 256, 0, stream>>>(s, d, n);
  };
  auto gemm = [&](const unsigned short* A, int lda, const unsigned short* W,
                  int ldw, float* C, int ldc, const float* bias, int Mtiles,
                  int Ntiles, int K) {
    int Ngrp = Ntiles / 4;  // all call sites have Ntiles % 4 == 0
    long waves = (long)Mtiles * Ngrp;
    int blocks = (int)((waves * 32 + 255) / 256);
    gemm_bf16_k<<<blocks, 256, 0, stream>>>(A, lda, W, ldw, C, ldc, bias,
                                            Mtiles, Ngrp, K);
  };

  // --- weight conversions (f32 -> bf16) ---
  cvt(enc_Wih,  WihB,   (long)4 * 1024 * 512);
  cvt(enc_Whh,  WhhB,   (long)4 * 1024 * 256);
  cvt(dec_Wih0, Wih0B,  (long)2048 * 1024);
  cvt(dec_Wih1, Wih1B,  (long)2048 * 512);
  cvt(dec_Whh,  WhhdB,  (long)2 * 2048 * 512);
  cvt(lininW,   lininB, (long)512 * 512);
  cvt(mixW,     mixB,   (long)512 * 1024);
  cvt(linoutW,  linoutB,(long)512 * 1536);
  cvt(stepdW,   stepdB, (long)512 * 1024);
  cvt(genW,     genB,   (long)32000 * 512);

  // --- embedding gathers ---
  {
    long n = (long)128 * 16 * 512;
    int blocks = (int)((n + 255) / 256);
    gather_enc_k<<<blocks, 256, 0, stream>>>(inp, encemb, xemb);
    gather_dec_k<<<blocks, 256, 0, stream>>>(outtok, decemb, demb);
    long nc = (long)16 * 64 * 512;
    gather_cat_k<<<(int)((nc + 255) / 256), 256, 0, stream>>>(catvec, catemb,
                                                              catencf);
  }

  // --- encoder layer 0 ---
  gemm(xemb, 512, WihB + (size_t)0 * 1024 * 512, 512, Gp0, 1024, nullptr, 128, 64, 512);
  gemm(xemb, 512, WihB + (size_t)1 * 1024 * 512, 512, Gp1, 1024, nullptr, 128, 64, 512);
  {
    size_t sm = (size_t)16 * 1024 * 4 + (size_t)16 * 256 * 4 + (size_t)16 * 256 * 2;
    enc_lstm_k<<<2, 512, sm, stream>>>(Gp0, Gp1, WhhB, enc_b, x1, nullptr,
                                       hfin, cfin);
  }
  // --- encoder layer 1 ---
  gemm(x1, 512, WihB + (size_t)2 * 1024 * 512, 512, Gp0, 1024, nullptr, 128, 64, 512);
  gemm(x1, 512, WihB + (size_t)3 * 1024 * 512, 512, Gp1, 1024, nullptr, 128, 64, 512);
  {
    size_t sm = (size_t)16 * 1024 * 4 + (size_t)16 * 256 * 4 + (size_t)16 * 256 * 2;
    enc_lstm_k<<<2, 512, sm, stream>>>(Gp0, Gp1, WhhB + (size_t)2 * 1024 * 256,
                                       enc_b + 2048, nullptr, encf,
                                       hfin + 8192, cfin + 8192);
  }

  // --- enc2cat (uses raw reshape of h0 [2,16,512] -> [16,1024]) ---
  cvt(hfin, hflatb, 16384);
  gemm(hflatb, 1024, stepdB, 1024, e2c, 512, nullptr, 1, 32, 1024);
  cvt(e2c, e2cb, (long)16 * 512);
  // catq_base = enc2cat @ encdecmix[:, :512]^T
  gemm(e2cb, 512, mixB, 1024, catqb, 512, nullptr, 1, 32, 512);

  // --- decoder input pre-GEMM: preDt = demb @ dec_Wih0[:, :512]^T ---
  gemm(demb, 512, Wih0B, 1024, preDt, 2048, nullptr, 128, 128, 512);

  // --- decoder scan (single persistent workgroup) ---
  float* attn_out = (float*)d_out + (long)16 * 128 * 32000;
  dec_k<<<1, 512, 294912, stream>>>(
      preDt, Wih0B, Wih1B, WhhdB, WhhdB + (size_t)2048 * 512, dec_bi, lininB,
      mixB, catqb, linoutB, encf, catencf, hfin, cfin, opsB, attn_out);

  // --- vocab logits: 2000 n-tiles x 2 t-groups ---
  logits_k<<<4000, 256, 0, stream>>>(opsB, genB, genb, (float*)d_out);
}
